// Attention_40991167873617
// MI455X (gfx1250) — matrix operand order repaired and checked by reading, not compiled, not run on hardware
//
#include <hip/hip_runtime.h>
#include <hip/hip_bf16.h>

typedef __attribute__((ext_vector_type(16))) __bf16 v16bf;
typedef __attribute__((ext_vector_type(8)))  float  v8f;
typedef __attribute__((ext_vector_type(4)))  int    v4i;

#define B_SZ 8192
#define T_SZ 64
#define E_SZ 256

#if __has_builtin(__builtin_amdgcn_global_load_async_to_lds_b128)
#define USE_ASYNC_LDS 1
typedef __attribute__((address_space(1))) v4i as1_v4i;
typedef __attribute__((address_space(3))) v4i as3_v4i;
#else
#define USE_ASYNC_LDS 0
#endif

// -------------------------------------------------------------------------
// Setup: convert attn_w (512x256 f32 row-major) into WMMA B-fragment-ordered
// bf16 in the workspace: fragment (n_tile, k_step) covers rows
// [k_step*32, +32) x cols [n_tile*16, +16); per-lane 16 halves contiguous
// (32 B) so the GEMM fetches a fragment with one wide load per lane.
// -------------------------------------------------------------------------
__global__ __launch_bounds__(256) void convert_w_kernel(
    const float* __restrict__ attn_w, unsigned short* __restrict__ w_frag) {
  int idx = blockIdx.x * 256 + threadIdx.x;   // 512*256 elements
  int kk = idx >> 8;                          // row 0..511
  int n  = idx & 255;                         // col 0..255
  float w = attn_w[idx];
  int n_tile = n >> 4;
  int kstep  = kk >> 5;
  int kr     = kk & 31;
  int laneHi = kr >> 4;
  int jr     = kr & 15;
  int lane   = (n & 15) | (laneHi << 4);
  size_t pos = ((size_t)((n_tile * 16 + kstep) * 32 + lane)) * 16 + jr;
  __bf16 h = (__bf16)w;
  w_frag[pos] = __builtin_bit_cast(unsigned short, h);
}

// Build a 16-bit A fragment from fp32 data at rp (k = 16*half + s, as in w_frag; caller
// passes rp already offset by the lane's koff0). Loads are issued as four
// independent b128s into distinct values so the scheduler can clause them
// and wait once, instead of a load->wait->cvt serial chain.
__device__ inline v16bf build_afrag(const float* rp) {
  float4 f0 = *(const float4*)(rp);
  float4 f1 = *(const float4*)(rp + 4);
  float4 f2 = *(const float4*)(rp + 8);
  float4 f3 = *(const float4*)(rp + 12);
  v16bf a;
  a[0]  = (__bf16)f0.x; a[1]  = (__bf16)f0.y;
  a[2]  = (__bf16)f0.z; a[3]  = (__bf16)f0.w;
  a[4]  = (__bf16)f1.x; a[5]  = (__bf16)f1.y;
  a[6]  = (__bf16)f1.z; a[7]  = (__bf16)f1.w;
  a[8]  = (__bf16)f2.x; a[9]  = (__bf16)f2.y;
  a[10] = (__bf16)f2.z; a[11] = (__bf16)f2.w;
  a[12] = (__bf16)f3.x; a[13] = (__bf16)f3.y;
  a[14] = (__bf16)f3.z; a[15] = (__bf16)f3.w;
  return a;
}

__device__ inline v16bf load_bfrag(const __bf16* w_frag, int frag_idx, int lane) {
  return *(const v16bf*)(w_frag + ((size_t)(frag_idx * 32 + lane)) * 16);
}

// -------------------------------------------------------------------------
// Main kernel: one workgroup (256 threads / 8 wave32) per batch element.
// Wave w owns n_tiles {2w, 2w+1} across ALL 4 m-tiles -> every W_bot
// fragment is loaded exactly once per block.
// -------------------------------------------------------------------------
__global__ __launch_bounds__(256) void attn_kernel(
    const float* __restrict__ x,
    const float* __restrict__ attn_b,
    const float* __restrict__ proj_w,
    const float* __restrict__ proj_b,
    const unsigned short* __restrict__ w_frag_u,
    float* __restrict__ out) {
  __shared__ float trees_f[T_SZ * E_SZ];  // 64 KB fp32 tile
  __shared__ float ep_sh[E_SZ];
  __shared__ float logits_sh[T_SZ];
  __shared__ float scores_sh[T_SZ];

  const int tid    = threadIdx.x;
  const int lane   = tid & 31;
  const int wave   = tid >> 5;
  const int laneHi = lane >> 4;
  const int lm     = lane & 15;
  const int b      = blockIdx.x;
  const __bf16* w_frag = (const __bf16*)w_frag_u;

  const float* xb = x + (size_t)b * (T_SZ + 2) * E_SZ;

  // ---- stage trees (64x256 f32 = 64 KB) into LDS ----
#if USE_ASYNC_LDS
  {
    // async DMA: overlaps with the ep/c-stage below (ASYNCcnt-tracked)
    const char* g = (const char*)(xb + 2 * E_SZ);
    char* l = (char*)trees_f;
#pragma unroll
    for (int i = 0; i < 16; ++i) {
      int off = (tid + i * 256) * 16;
      __builtin_amdgcn_global_load_async_to_lds_b128(
          (as1_v4i*)(g + off), (as3_v4i*)(l + off), 0, 0);
    }
  }
#else
  {
    const float4* src = (const float4*)(xb + 2 * E_SZ);
    float4* dst = (float4*)trees_f;
#pragma unroll
    for (int i = 0; i < 16; ++i) dst[tid + i * 256] = src[tid + i * 256];
  }
#endif

  // ---- ep = x[b,0]*x[b,1]; also the second output ----
  {
    float ep = xb[tid] * xb[E_SZ + tid];
    ep_sh[tid] = ep;
    out[(size_t)B_SZ * E_SZ + (size_t)b * E_SZ + tid] = ep;
  }
  if (tid < T_SZ) logits_sh[tid] = proj_b[0];
  __syncthreads();  // ep_sh ready (trees ready too on the sync path)

  const int koff0 = laneHi ? 16 : 0;  // A-layout offset for hi half-wave
  const int nt0 = wave * 2, nt1 = wave * 2 + 1;

  // ---- c = ep @ W_top + attn_b via WMMA; result stays in registers ----
  float cv0, cv1;
  {
    v8f c0, c1;
    float b0 = attn_b[nt0 * 16 + lm];
    float b1 = attn_b[nt1 * 16 + lm];
#pragma unroll
    for (int r = 0; r < 8; ++r) { c0[r] = b0; c1[r] = b1; }
#pragma unroll
    for (int ks = 0; ks < 8; ++ks) {
      v16bf a = build_afrag(&ep_sh[ks * 32 + koff0]);
      v16bf w0 = load_bfrag(w_frag, nt0 * 16 + ks, lane);
      v16bf w1 = load_bfrag(w_frag, nt1 * 16 + ks, lane);
      c0 = __builtin_amdgcn_wmma_f32_16x16x32_bf16(false, a, false, w0,
                                                   (short)0, c0, false, false);
      c1 = __builtin_amdgcn_wmma_f32_16x16x32_bf16(false, a, false, w1,
                                                   (short)0, c1, false, false);
    }
    // every lane needs c[n_base + lm]; row M=0 lives in VGPR0 of lanes 0-15
    cv0 = __shfl(c0[0], lm, 32);
    cv1 = __shfl(c1[0], lm, 32);
  }

#if USE_ASYNC_LDS
#if __has_builtin(__builtin_amdgcn_s_wait_asynccnt)
  __builtin_amdgcn_s_wait_asynccnt(0);
#else
  asm volatile("s_wait_asynccnt 0x0" ::: "memory");
#endif
#endif
  __syncthreads();  // all waves' tree tiles are in LDS

  // ---- H = trees @ W_bot + c ; fuse relu + proj_w dot into logits ----
  {
    v8f acc[4][2];
#pragma unroll
    for (int m = 0; m < 4; ++m)
#pragma unroll
      for (int r = 0; r < 8; ++r) { acc[m][0][r] = cv0; acc[m][1][r] = cv1; }

#pragma unroll
    for (int ks = 0; ks < 8; ++ks) {
      v16bf w0 = load_bfrag(w_frag, nt0 * 16 + 8 + ks, lane);
      v16bf w1 = load_bfrag(w_frag, nt1 * 16 + 8 + ks, lane);
#pragma unroll
      for (int m = 0; m < 4; ++m) {
        v16bf a = build_afrag(&trees_f[(m * 16 + lm) * E_SZ + ks * 32 + koff0]);
        acc[m][0] = __builtin_amdgcn_wmma_f32_16x16x32_bf16(
            false, a, false, w0, (short)0, acc[m][0], false, false);
        acc[m][1] = __builtin_amdgcn_wmma_f32_16x16x32_bf16(
            false, a, false, w1, (short)0, acc[m][1], false, false);
      }
    }

    float pw0 = proj_w[nt0 * 16 + lm];
    float pw1 = proj_w[nt1 * 16 + lm];
#pragma unroll
    for (int m = 0; m < 4; ++m) {
#pragma unroll
      for (int r = 0; r < 8; ++r) {
        float h0 = acc[m][0][r] > 0.0f ? acc[m][0][r] : 0.0f;  // relu
        float h1 = acc[m][1][r] > 0.0f ? acc[m][1][r] : 0.0f;
        float v = h0 * pw0 + h1 * pw1;
        v += __shfl_xor(v, 1, 32);
        v += __shfl_xor(v, 2, 32);
        v += __shfl_xor(v, 4, 32);
        v += __shfl_xor(v, 8, 32);
        if (lm == 0)  // lane 0: rows M=r, lane 16: rows M=r+8
          atomicAdd(&logits_sh[m * 16 + r + (laneHi ? 8 : 0)], v);
      }
    }
  }
  __syncthreads();

  // ---- softmax over T=64 (wave 0), 1/num_tree folded into scores ----
  if (wave == 0) {
    float l0 = logits_sh[lane];
    float l1 = logits_sh[lane + 32];
    float m = fmaxf(l0, l1);
#pragma unroll
    for (int mask = 16; mask >= 1; mask >>= 1)
      m = fmaxf(m, __shfl_xor(m, mask, 32));
    float e0 = __expf(l0 - m);
    float e1 = __expf(l1 - m);
    float s = e0 + e1;
#pragma unroll
    for (int mask = 16; mask >= 1; mask >>= 1)
      s += __shfl_xor(s, mask, 32);
    float inv = 1.0f / (s * (float)T_SZ);
    scores_sh[lane]      = e0 * inv;
    scores_sh[lane + 32] = e1 * inv;
  }
  __syncthreads();

  // ---- attn_output[b, e] = sum_t score[t] * trees[t, e] (fp32) ----
  {
    float acc = 0.0f;
#pragma unroll
    for (int t = 0; t < T_SZ; ++t)
      acc += scores_sh[t] * trees_f[t * E_SZ + tid];
    out[(size_t)b * E_SZ + tid] = acc;
  }
}

extern "C" void kernel_launch(void* const* d_in, const int* in_sizes, int n_in,
                              void* d_out, int out_size, void* d_ws, size_t ws_size,
                              hipStream_t stream) {
  (void)in_sizes; (void)n_in; (void)out_size; (void)ws_size;
  const float* x      = (const float*)d_in[0];
  const float* attn_w = (const float*)d_in[1];
  const float* attn_b = (const float*)d_in[2];
  const float* proj_w = (const float*)d_in[3];
  const float* proj_b = (const float*)d_in[4];
  float* out = (float*)d_out;
  unsigned short* w_frag = (unsigned short*)d_ws;  // 512*256 bf16 = 256 KB

  convert_w_kernel<<<512, 256, 0, stream>>>(attn_w, w_frag);
  attn_kernel<<<B_SZ, 256, 0, stream>>>(x, attn_b, proj_w, proj_b, w_frag, out);
}
